// LPLayer_23570780520895
// MI455X (gfx1250) — matrix operand order repaired and checked by reading, not compiled, not run on hardware
//
#include <hip/hip_runtime.h>

typedef __bf16 bf16_t;
typedef __attribute__((ext_vector_type(16))) __bf16 v16bf;
typedef __attribute__((ext_vector_type(8)))  __bf16 v8bf;
typedef __attribute__((ext_vector_type(4)))  __bf16 v4bf;
typedef __attribute__((ext_vector_type(8)))  float  v8f;
typedef __attribute__((ext_vector_type(4)))  int    v4i;

#define LDS_K 264  // 256 + 8 bf16 pad; row stride 528B = 33*16 stays 16B-aligned

#if __has_builtin(__builtin_amdgcn_global_load_async_to_lds_b128)
#define HAVE_ASYNC_LDS 1
#endif

// 16-byte global->LDS copy helper (async DMA when available)
__device__ __forceinline__ void cp16_g2l(const bf16_t* __restrict__ g,
                                         bf16_t* __restrict__ l)
{
#ifdef HAVE_ASYNC_LDS
  __builtin_amdgcn_global_load_async_to_lds_b128(
      (__attribute__((address_space(1))) v4i*)(void*)g,
      (__attribute__((address_space(3))) v4i*)(void*)l, 0, 0);
#else
  *(uint4*)l = *(const uint4*)g;
#endif
}

__device__ __forceinline__ void wait_async_lds()
{
#ifdef HAVE_ASYNC_LDS
#if __has_builtin(__builtin_amdgcn_s_wait_asynccnt)
  __builtin_amdgcn_s_wait_asynccnt(0);
#else
  asm volatile("s_wait_asynccnt 0" ::: "memory");
#endif
#endif
}

// ---------------------------------------------------------------------------
// Prepass A: hB = bf16(h)   (one conversion instead of 4x inside the GEMM)
// ---------------------------------------------------------------------------
__global__ void gat_cvt_h(const float* __restrict__ h, bf16_t* __restrict__ hB,
                          long long n4)  // n4 = N*256/4
{
  long long i = (long long)blockIdx.x * blockDim.x + threadIdx.x;
  long long stride = (long long)gridDim.x * blockDim.x;
  const float4* h4 = (const float4*)h;
  for (; i < n4; i += stride) {
    float4 v = h4[i];
    v4bf o = {(bf16_t)v.x, (bf16_t)v.y, (bf16_t)v.z, (bf16_t)v.w};
    *(v4bf*)(hB + i * 4) = o;
  }
}

// ---------------------------------------------------------------------------
// Prepass B: WT[n][k] = bf16(W[k][n])   (pre-transposed, 128 KB, L2-resident)
// ---------------------------------------------------------------------------
__global__ void gat_cvt_wt(const float* __restrict__ W, bf16_t* __restrict__ WT)
{
  int idx = blockIdx.x * 256 + threadIdx.x;   // 65536 threads
  int n = idx >> 8;
  int k = idx & 255;
  WT[n * 256 + k] = (bf16_t)W[k * 256 + n];
}

// ---------------------------------------------------------------------------
// Kernel 1: Wh = h @ W  via v_wmma_f32_16x16x32_bf16
// block = 256 threads (8 waves). Block tile: 32 rows x 64 cols.
// A/B tiles are bf16 in global (hB, WT) -> async-DMA'd straight into LDS.
// ---------------------------------------------------------------------------
__global__ __launch_bounds__(256) void gat_gemm_wmma(
    const bf16_t* __restrict__ hB, const bf16_t* __restrict__ WT,
    float* __restrict__ Wh)
{
  __shared__ bf16_t Alds[32][LDS_K];   // Alds[m][k] = hB[m_block+m][k]
  __shared__ bf16_t Wlds[64][LDS_K];   // Wlds[n][k] = WT[n_block+n][k]

  const int t       = threadIdx.x;
  const int lane    = t & 31;
  const int wave    = t >> 5;
  const int m_block = blockIdx.x * 32;
  const int n_block = blockIdx.y * 64;

  // --- stage A tile: 32 rows x 512B; one wave copies a row per op ---
  {
    const bf16_t* gA = hB + (size_t)m_block * 256;
    #pragma unroll
    for (int i = 0; i < 4; ++i) {
      int row = wave + i * 8;
      cp16_g2l(gA + row * 256 + lane * 8, &Alds[row][lane * 8]);
    }
  }
  // --- stage W tile: 64 rows x 512B of pre-transposed WT ---
  {
    const bf16_t* gB = WT + (size_t)n_block * 256;
    #pragma unroll
    for (int i = 0; i < 8; ++i) {
      int row = wave + i * 8;
      cp16_g2l(gB + row * 256 + lane * 8, &Wlds[row][lane * 8]);
    }
  }
  wait_async_lds();
  __syncthreads();

  const int rg   = wave >> 2;    // 0..1  (16-row group)
  const int ct   = wave & 3;     // 0..3  (16-col tile)
  const int half = lane >> 4;    // 0 or 1
  const int l15  = lane & 15;
  const int arow = rg * 16 + l15;
  const int brow = ct * 16 + l15;

  v8f acc = {};
  #pragma unroll
  for (int kk = 0; kk < 8; ++kk) {
    const int k0 = kk * 32;
    // A fragment (16-bit A 16x32 layout): lane<16 -> K[k0..k0+7],K[k0+16..k0+23]
    //                                     lane>=16 -> +8 shifted halves
    const int ka = k0 + half * 8;
    v8bf alo = *(const v8bf*)&Alds[arow][ka];
    v8bf ahi = *(const v8bf*)&Alds[arow][ka + 16];
    v16bf a = __builtin_shufflevector(alo, ahi,
        0,1,2,3,4,5,6,7,8,9,10,11,12,13,14,15);
    // B fragment (32x16): same K order as A: K[kb..kb+7], K[kb+16..kb+23] for column l15
    const int kb = k0 + half * 8;
    v8bf blo = *(const v8bf*)&Wlds[brow][kb];
    v8bf bhi = *(const v8bf*)&Wlds[brow][kb + 16];
    v16bf b = __builtin_shufflevector(blo, bhi,
        0,1,2,3,4,5,6,7,8,9,10,11,12,13,14,15);
    acc = __builtin_amdgcn_wmma_f32_16x16x32_bf16(
        false, a, false, b, (short)0, acc, false, false);
  }

  // C/D layout: VGPR j -> row half*8+j, col = lane&15
  const int col   = n_block + ct * 16 + l15;
  const int rbase = m_block + rg * 16 + half * 8;
  #pragma unroll
  for (int j = 0; j < 8; ++j)
    Wh[(size_t)(rbase + j) * 256 + col] = acc[j];
}

// ---------------------------------------------------------------------------
// Kernel 0: zero a float buffer (h_prime accumulator + row_sum)
// ---------------------------------------------------------------------------
__global__ void gat_zero(float* __restrict__ p, long long n)
{
  long long i = (long long)blockIdx.x * blockDim.x + threadIdx.x;
  long long stride = (long long)gridDim.x * blockDim.x;
  for (; i < n; i += stride) p[i] = 0.0f;
}

// ---------------------------------------------------------------------------
// Kernel 2: e = leakyrelu(<label[src],label[dst]>); exp; atomic row_sum[src]
// ---------------------------------------------------------------------------
__global__ __launch_bounds__(256) void gat_edge_logits(
    const float* __restrict__ label, const int* __restrict__ src,
    const int* __restrict__ dst, float* __restrict__ exp_e,
    float* __restrict__ row_sum, int E)
{
  int e = blockIdx.x * 256 + threadIdx.x;
  if (e >= E) return;
  const int s = src[e];
  const int d = dst[e];
  const float4* ls = (const float4*)label + (size_t)s * 8;
  const float4* ld = (const float4*)label + (size_t)d * 8;
  float dot = 0.0f;
  #pragma unroll
  for (int i = 0; i < 8; ++i) {
    float4 a = ls[i];
    float4 b = ld[i];
    dot += a.x * b.x + a.y * b.y + a.z * b.z + a.w * b.w;
  }
  float lr = (dot >= 0.0f) ? dot : 0.2f * dot;
  float ee = __expf(lr);
  exp_e[e] = ee;
  unsafeAtomicAdd(&row_sum[s], ee);  // HW global_atomic_add_f32
}

// ---------------------------------------------------------------------------
// Kernel 3: h_prime[src] += (exp_e / row_sum[src]) * Wh[dst]
// One wave per edge iteration: 32 lanes x 8 floats = 256-wide row.
// ---------------------------------------------------------------------------
__global__ __launch_bounds__(256) void gat_scatter(
    const float* __restrict__ Wh, const float* __restrict__ exp_e,
    const float* __restrict__ row_sum, const int* __restrict__ src,
    const int* __restrict__ dst, float* __restrict__ out, int E)
{
  const int lane = threadIdx.x & 31;
  const int wid  = (blockIdx.x * blockDim.x + threadIdx.x) >> 5;
  const int nw   = (gridDim.x * blockDim.x) >> 5;
  for (int e = wid; e < E; e += nw) {
    const int s = src[e];
    const int d = dst[e];
    const float att = exp_e[e] / fmaxf(row_sum[s], 1e-9f);
    const float4* w4 = (const float4*)(Wh + (size_t)d * 256) + lane * 2;
    float4 a = w4[0];
    float4 b = w4[1];
    float* o = out + (size_t)s * 256 + lane * 8;
    unsafeAtomicAdd(o + 0, att * a.x);
    unsafeAtomicAdd(o + 1, att * a.y);
    unsafeAtomicAdd(o + 2, att * a.z);
    unsafeAtomicAdd(o + 3, att * a.w);
    unsafeAtomicAdd(o + 4, att * b.x);
    unsafeAtomicAdd(o + 5, att * b.y);
    unsafeAtomicAdd(o + 6, att * b.z);
    unsafeAtomicAdd(o + 7, att * b.w);
  }
}

// ---------------------------------------------------------------------------
extern "C" void kernel_launch(void* const* d_in, const int* in_sizes, int n_in,
                              void* d_out, int out_size, void* d_ws, size_t ws_size,
                              hipStream_t stream)
{
  const float* h     = (const float*)d_in[0];  // [N, 256]
  const float* label = (const float*)d_in[1];  // [N, 32]
  const float* W     = (const float*)d_in[2];  // [256, 256]
  const int*   adj   = (const int*)d_in[3];    // [2, E] (int32)

  const int N = in_sizes[0] / 256;             // 100000
  const int E = in_sizes[3] / 2;               // 1600000
  const int* src = adj;
  const int* dst = adj + E;
  float* out = (float*)d_out;                  // [N, 256]

  // workspace: Wh [N*256 f32] | exp_e [E f32] | row_sum [N f32]
  //            | hB [N*256 bf16] | WT [256*256 bf16]
  float*  Wh      = (float*)d_ws;
  float*  exp_e   = Wh + (size_t)N * 256;
  float*  row_sum = exp_e + (size_t)E;
  bf16_t* hB      = (bf16_t*)(row_sum + N);
  bf16_t* WT      = hB + (size_t)N * 256;

  // zero accumulators (harness poisons them; atomics need clean zeros)
  gat_zero<<<2048, 256, 0, stream>>>(out, (long long)N * 256);
  gat_zero<<<256, 256, 0, stream>>>(row_sum, (long long)N);

  // bf16 prepass: hB = bf16(h), WT = bf16(W^T)
  gat_cvt_h<<<4096, 256, 0, stream>>>(h, hB, (long long)N * 64);
  gat_cvt_wt<<<256, 256, 0, stream>>>(W, WT);

  // Wh = h @ W   (bf16 WMMA, f32 accumulate, async LDS staging)
  gat_gemm_wmma<<<dim3(N / 32, 4), 256, 0, stream>>>(hB, WT, Wh);

  // edge logits + segment-softmax denominator
  gat_edge_logits<<<(E + 255) / 256, 256, 0, stream>>>(
      label, src, dst, exp_e, row_sum, E);

  // weighted scatter-aggregate
  gat_scatter<<<2048, 256, 0, stream>>>(Wh, exp_e, row_sum, src, dst, out, E);
}
